// MTPForCausalLM_25013889532222
// MI455X (gfx1250) — matrix operand order repaired and checked by reading, not compiled, not run on hardware
//
#include <hip/hip_runtime.h>
#include <hip/hip_bf16.h>
#include <stdint.h>

typedef __attribute__((ext_vector_type(16))) __bf16 v16bf;
typedef __attribute__((ext_vector_type(8)))  __bf16 v8bf;
typedef __attribute__((ext_vector_type(4)))  __bf16 v4bf;
typedef __attribute__((ext_vector_type(8)))  float  v8f;

#define HID   7168
#define NTOK  8192
#define NEXP  256
#define MT    64          // tokens per block
#define KS    32          // k-step
#define ASTR  40          // padded LDS row stride (bf16 elems) -> conflict-free b128 reads
#define GEMM_THREADS 512  // 16 waves

// split fp32 -> bf16 hi + bf16 lo (residual), RNE
__device__ __forceinline__ void bsplit(float x, __bf16 &h, __bf16 &l) {
  h = (__bf16)x;
  l = (__bf16)(x - (float)h);
}

// ---------------------------------------------------------------------------
// Kernel 1: router GEMM, logits[T,E] = hidden[T,H] @ weight[E,H]^T
// bf16x3 emulation of fp32 on v_wmma_f32_16x16x32_bf16, fp32 accumulate.
// Block: 512 thr = 16 waves; wave w -> experts [16w,16w+16); 4 M-subtiles of 16.
// ---------------------------------------------------------------------------
__global__ __launch_bounds__(GEMM_THREADS)
void router_gemm(const float* __restrict__ hs, const float* __restrict__ wt,
                 float* __restrict__ logits) {
  __shared__ __bf16 Ah[MT][ASTR];
  __shared__ __bf16 Al[MT][ASTR];

  const int tid  = threadIdx.x;
  const int wave = tid >> 5;
  const int lane = tid & 31;
  const int half = lane >> 4;   // 0/1: K-half selector
  const int lr   = lane & 15;
  const int m0   = blockIdx.x * MT;
  const int n0   = wave * 16;

  v8f acc[4] = {};

  // cooperative A staging: thread -> (row = tid/8, 4 consecutive k = (tid%8)*4)
  const int arow = tid >> 3;
  const int acol = (tid & 7) * 4;
  const float* aptr = hs + (size_t)(m0 + arow) * HID + acol;
  // B fragment: lane -> column n0+lr, K in {kb..kb+8} U {kb+16..kb+24}, kb = half*8
  const float* bptr = wt + (size_t)(n0 + lr) * HID + half * 8;

  for (int k = 0; k < HID; k += KS) {
    // ---- global loads (fp32) ----
    float4 av = *(const float4*)(aptr + k);
    float4 b0 = *(const float4*)(bptr + k);
    float4 b1 = *(const float4*)(bptr + k + 4);
    float4 b2 = *(const float4*)(bptr + k + 16);
    float4 b3 = *(const float4*)(bptr + k + 20);

    // ---- split B into hi/lo bf16 fragments (per-wave private) ----
    float bf[16] = { b0.x,b0.y,b0.z,b0.w, b1.x,b1.y,b1.z,b1.w,
                     b2.x,b2.y,b2.z,b2.w, b3.x,b3.y,b3.z,b3.w };
    v16bf bh, bl;
    #pragma unroll
    for (int j = 0; j < 16; ++j) {
      __bf16 h, l; bsplit(bf[j], h, l);
      bh[j] = h; bl[j] = l;
    }

    // ---- stage A hi/lo into LDS (shared by all 16 waves) ----
    __syncthreads();                       // previous-iter fragment reads done
    {
      float af[4] = { av.x, av.y, av.z, av.w };
      v4bf h4, l4;
      #pragma unroll
      for (int j = 0; j < 4; ++j) {
        __bf16 h, l; bsplit(af[j], h, l);
        h4[j] = h; l4[j] = l;
      }
      *(v4bf*)&Ah[arow][acol] = h4;        // ds_store_b64, 8B aligned
      *(v4bf*)&Al[arow][acol] = l4;
    }
    __syncthreads();                       // stores visible

    // ---- 4 M-subtiles x (hi*hi + hi*lo + lo*hi) ----
    #pragma unroll
    for (int ms = 0; ms < 4; ++ms) {
      const int mr = ms * 16 + lr;
      // A frag: K in {kb..kb+8} U {kb+16..kb+24}, kb = half*8 (ISA 16-bit A layout)
      v8bf h0 = *(const v8bf*)&Ah[mr][half * 8];        // 16B aligned (stride 80B)
      v8bf h1 = *(const v8bf*)&Ah[mr][half * 8 + 16];
      v8bf l0 = *(const v8bf*)&Al[mr][half * 8];
      v8bf l1 = *(const v8bf*)&Al[mr][half * 8 + 16];
      v16bf ah = __builtin_shufflevector(h0, h1, 0,1,2,3,4,5,6,7,8,9,10,11,12,13,14,15);
      v16bf al = __builtin_shufflevector(l0, l1, 0,1,2,3,4,5,6,7,8,9,10,11,12,13,14,15);

      acc[ms] = __builtin_amdgcn_wmma_f32_16x16x32_bf16(false, ah, false, bh,
                                                        (short)0, acc[ms], false, false);
      acc[ms] = __builtin_amdgcn_wmma_f32_16x16x32_bf16(false, ah, false, bl,
                                                        (short)0, acc[ms], false, false);
      acc[ms] = __builtin_amdgcn_wmma_f32_16x16x32_bf16(false, al, false, bh,
                                                        (short)0, acc[ms], false, false);
    }
  }

  // C/D layout: lane (half,lr): N = lr, M = v + 8*half
  #pragma unroll
  for (int ms = 0; ms < 4; ++ms) {
    #pragma unroll
    for (int v = 0; v < 8; ++v) {
      logits[(size_t)(m0 + ms * 16 + half * 8 + v) * NEXP + n0 + lr] = acc[ms][v];
    }
  }
}

// ---------------------------------------------------------------------------
// Kernel 2: noaux_tc routing, one wave32 per token, lane l owns experts
// [8l, 8l+8) (entirely inside group l>>2 since groups are 32 experts).
// ---------------------------------------------------------------------------
__global__ __launch_bounds__(256)
void router_topk(const float* __restrict__ logits, const float* __restrict__ bias,
                 float* __restrict__ ovals, int* __restrict__ oidx) {
  const int lane = threadIdx.x & 31;
  const int t    = blockIdx.x * 8 + (threadIdx.x >> 5);
  const float* lrow = logits + (size_t)t * NEXP;
  const float NEGINF = -__builtin_inff();

  float s[8], swb[8];
  #pragma unroll
  for (int j = 0; j < 8; ++j) {
    const int e = lane * 8 + j;
    float sg = 1.0f / (1.0f + __expf(-lrow[e]));
    s[j]   = sg;
    swb[j] = sg + bias[e];
  }

  // group score = sum of top-2 biased scores within the 32-expert group
  float m1 = NEGINF, m2 = NEGINF;
  #pragma unroll
  for (int j = 0; j < 8; ++j) {
    float v = swb[j];
    if (v > m1) { m2 = m1; m1 = v; } else if (v > m2) { m2 = v; }
  }
  #pragma unroll
  for (int off = 1; off <= 2; off <<= 1) {   // merge across the 4 lanes of a group
    float o1 = __shfl_xor(m1, off, 32);
    float o2 = __shfl_xor(m2, off, 32);
    float nm1 = fmaxf(m1, o1);
    float nm2 = fmaxf(fminf(m1, o1), fmaxf(m2, o2));
    m1 = nm1; m2 = nm2;
  }
  const float gs = m1 + m2;

  float gall[8];
  #pragma unroll
  for (int g = 0; g < 8; ++g) gall[g] = __shfl(gs, g * 4, 32);

  // top-4 groups (ties -> lower index, jax top_k semantics)
  unsigned gmask = 0;
  #pragma unroll
  for (int p = 0; p < 4; ++p) {
    float best = NEGINF; int bg = 0;
    #pragma unroll
    for (int g = 0; g < 8; ++g)
      if (!((gmask >> g) & 1) && gall[g] > best) { best = gall[g]; bg = g; }
    gmask |= 1u << bg;
  }

  // masked top-8 experts: masked-out groups contribute exactly 0.0 (as reference)
  const bool gsel = (gmask >> (lane >> 2)) & 1;
  unsigned picked = 0;
  float selS[8]; int selI[8];
  for (int p = 0; p < 8; ++p) {
    float bv = NEGINF; int bi = 1 << 30; float bs = 0.0f;
    #pragma unroll
    for (int j = 0; j < 8; ++j) {
      if ((picked >> j) & 1) continue;
      float v = gsel ? swb[j] : 0.0f;
      if (v > bv) { bv = v; bi = lane * 8 + j; bs = s[j]; }
    }
    #pragma unroll
    for (int off = 16; off >= 1; off >>= 1) {   // wave argmax, index asc tie-break
      float ov = __shfl_xor(bv, off, 32);
      int   oi = __shfl_xor(bi, off, 32);
      float os = __shfl_xor(bs, off, 32);
      if (ov > bv || (ov == bv && oi < bi)) { bv = ov; bi = oi; bs = os; }
    }
    selS[p] = bs; selI[p] = bi;
    if ((bi >> 3) == lane) picked |= 1u << (bi & 7);
  }

  // renormalize UNBIASED sigmoid scores, scale by 2.5
  float sum = 1e-20f;
  #pragma unroll
  for (int p = 0; p < 8; ++p) sum += selS[p];
  const float sc = 2.5f / sum;
  #pragma unroll
  for (int p = 0; p < 8; ++p) selS[p] *= sc;

  // final top_k: sort 8 desc by (value, index asc)
  for (int a = 1; a < 8; ++a) {
    float v = selS[a]; int i = selI[a];
    int b = a - 1;
    while (b >= 0 && (selS[b] < v || (selS[b] == v && selI[b] > i))) {
      selS[b + 1] = selS[b]; selI[b + 1] = selI[b]; --b;
    }
    selS[b + 1] = v; selI[b + 1] = i;
  }

  if (lane < 8) {
    ovals[(size_t)t * 8 + lane] = selS[lane];
    oidx [(size_t)t * 8 + lane] = selI[lane];
  }
}

// ---------------------------------------------------------------------------
extern "C" void kernel_launch(void* const* d_in, const int* in_sizes, int n_in,
                              void* d_out, int out_size, void* d_ws, size_t ws_size,
                              hipStream_t stream) {
  (void)in_sizes; (void)n_in; (void)out_size; (void)ws_size;
  const float* hs   = (const float*)d_in[0];   // [8192, 7168] fp32
  const float* wt   = (const float*)d_in[1];   // [256, 7168]  fp32
  const float* bias = (const float*)d_in[2];   // [256]        fp32

  float* logits = (float*)d_ws;                // 8192*256*4 = 8 MB scratch
  float* ovals  = (float*)d_out;               // [8192,8] fp32 values
  int*   oidx   = (int*)d_out + (size_t)NTOK * 8;  // [8192,8] int32 indices

  router_gemm<<<NTOK / MT, GEMM_THREADS, 0, stream>>>(hs, wt, logits);
  router_topk<<<NTOK / 8, 256, 0, stream>>>(logits, bias, ovals, oidx);
}
